// NPChangeTransitionPrior_69157563400603
// MI455X (gfx1250) — matrix operand order repaired and checked by reading, not compiled, not run on hardware
//
#include <hip/hip_runtime.h>

typedef __attribute__((ext_vector_type(16))) _Float16 v16h;
typedef __attribute__((ext_vector_type(8)))  float    v8f;

#define LAGS   2
#define LATENT 8
#define HID    64
#define TLEN   2048
#define LEN    2046
#define NBATCH 64
#define NSAMP  (NBATCH * LEN)   // 130944 = 64 rows/block * 2046 blocks

// ---- LDS layout (halves) ----
// WT1: [64][96]  transposed L1 weights (Kpad=96)         @ 0      (6144)
// WT2: [64][64]  transposed L2 weights                   @ 6144   (4096)
// WT3: [64][64]  transposed L3 weights                   @ 10240  (4096)
// WT4: [64][64]  transposed L4 weights (g: [16][64])     @ 14336  (4096)
// GIN: 4 waves x [16][96]  g-input staging               @ 18432  (6144)
// HST: 4 waves x [16][64]  h / dh staging                @ 24576  (4096)
#define LDS_HALVES 28672

static __device__ inline v8f vzero8() {
  v8f v;
#pragma unroll
  for (int i = 0; i < 8; ++i) v[i] = 0.0f;
  return v;
}

// A-fragment (16x32 f16) gather from row-major LDS [16][ld] at k-offset k0.
// Layout: lane m=l&15; lanes<16 hold K k0+0..15,
// lanes>=16 hold K k0+16..31. One contiguous 16-half run per lane.
static __device__ inline v16h lds_load_a(const _Float16* base, int ld, int k0, int lane) {
  int m = lane & 15;
  int hi8 = (lane >> 4) << 4;
  const _Float16* p = base + m * ld + k0 + hi8;
  v16h a;
#pragma unroll
  for (int i = 0; i < 8; ++i) a[i] = p[i];
#pragma unroll
  for (int i = 0; i < 8; ++i) a[8 + i] = p[8 + i];
  return a;
}

// B-fragment (32x16 f16) gather from TRANSPOSED LDS weights Wt[N][ldK]:
// lane holds column n=n0+(l&15), K run k0+16*(l>>4) .. +15 (contiguous).
static __device__ inline v16h lds_load_bT(const _Float16* Wt, int ldK, int k0, int n0, int lane) {
  int n  = n0 + (lane & 15);
  int kk = k0 + ((lane >> 4) << 4);
  const _Float16* p = Wt + n * ldK + kk;
  v16h b;
#pragma unroll
  for (int i = 0; i < 16; ++i) b[i] = p[i];
  return b;
}

// C/D tile element (r) maps to (m = r + 8*(l>>4), n = l&15) per ISA 7.12.2.
static __device__ inline void stage_c(_Float16* dst /*[16][64]*/, const v8f c[4], int lane) {
  int n = lane & 15;
  int m0 = (lane >> 4) << 3;
#pragma unroll
  for (int nt = 0; nt < 4; ++nt)
#pragma unroll
    for (int r = 0; r < 8; ++r)
      dst[(m0 + r) * 64 + nt * 16 + n] = (_Float16)c[nt][r];
}

// 16xK(ch*32) @ Kx64 -> accumulate into 4 16x16 tiles; A staged in LDS rows.
template <int KCH>
static __device__ inline void wave_gemm(const _Float16* Ast, int ldA,
                                        const _Float16* Wt, int ldK,
                                        v8f c[4], int lane) {
#pragma unroll
  for (int kc = 0; kc < KCH; ++kc) {
    v16h a = lds_load_a(Ast, ldA, kc * 32, lane);
#pragma unroll
    for (int nt = 0; nt < 4; ++nt) {
      v16h b = lds_load_bT(Wt, ldK, kc * 32, nt * 16, lane);
      c[nt] = __builtin_amdgcn_wmma_f32_16x16x32_f16(false, a, false, b, (short)0, c[nt], false, false);
    }
  }
}

// Same but A fragments (K=64 -> 2 chunks) already in registers.
static __device__ inline void frag_gemm(v16h a0, v16h a1, const _Float16* Wt, int ldK,
                                        v8f c[4], int lane) {
#pragma unroll
  for (int nt = 0; nt < 4; ++nt) {
    v16h b0 = lds_load_bT(Wt, ldK, 0, nt * 16, lane);
    c[nt] = __builtin_amdgcn_wmma_f32_16x16x32_f16(false, a0, false, b0, (short)0, c[nt], false, false);
    v16h b1 = lds_load_bT(Wt, ldK, 32, nt * 16, lane);
    c[nt] = __builtin_amdgcn_wmma_f32_16x16x32_f16(false, a1, false, b1, (short)0, c[nt], false, false);
  }
}

// Cooperative: global row-major W[K][N=64] (f32) -> LDS transposed Wt[64][Kpad] (f16), zero-pad K.
static __device__ inline void load_wt(const float* __restrict__ W, _Float16* Wt,
                                      int K, int Kpad, int tid) {
  for (int idx = tid; idx < 64 * Kpad; idx += 128) {
    int n = idx / Kpad;
    int k = idx - n * Kpad;
    Wt[idx] = (k < K) ? (_Float16)W[k * 64 + n] : (_Float16)0.0f;
  }
}

static __device__ inline void load_bias(const float* __restrict__ b, float* dst, int n, int tid) {
  for (int i = tid; i < n; i += 128) dst[i] = b[i];
}

__global__ __launch_bounds__(128) void np_flow_kernel(
    const float* __restrict__ x, const float* __restrict__ emb,
    const float* __restrict__ fcW1, const float* __restrict__ fcb1,
    const float* __restrict__ fcWh, const float* __restrict__ fcbh,
    const float* __restrict__ fcW2, const float* __restrict__ fcb2,
    const float* __restrict__ gW1, const float* __restrict__ gb1,
    const float* __restrict__ gWh, const float* __restrict__ gbh,
    const float* __restrict__ gW2, const float* __restrict__ gb2,
    float* __restrict__ out) {
  __shared__ _Float16 sh[LDS_HALVES];
  __shared__ float sbias[256];

  const int tid  = threadIdx.x;
  const int lane = tid & 31;
  const int wv   = tid >> 5;
  const int s0   = blockIdx.x * 64 + wv * 16;

  _Float16* WT1 = sh;
  _Float16* WT2 = sh + 6144;
  _Float16* WT3 = sh + 10240;
  _Float16* WT4 = sh + 14336;
  _Float16* GIN = sh + 18432 + wv * (16 * 96);
  _Float16* HST = sh + 24576 + wv * (16 * 64);

  // ---------------- Phase F: fc MLP on embeddings ----------------
  load_wt(fcW1, WT1, 8, 96, tid);
  load_wt(fcWh, WT2, 64, 64, tid);
  load_wt(fcWh + 4096, WT3, 64, 64, tid);
  load_wt(fcW2, WT4, 64, 64, tid);
  load_bias(fcb1, sbias, 64, tid);
  load_bias(fcbh, sbias + 64, 64, tid);
  load_bias(fcbh + 64, sbias + 128, 64, tid);
  load_bias(fcb2, sbias + 192, 64, tid);
  __syncthreads();

  // Stage wave inputs: emb rows -> HST cols 0..7 (pad 8..31 = 0);
  // x lags -> GIN cols 64..79; zero GIN cols 81..95.
  {
    int m = lane & 15;
    int s = s0 + m;
    int bb = s / LEN;
    int ii = s - bb * LEN;
    if (lane < 16) {
      const float* er = emb + (bb * TLEN + ii + LAGS) * LATENT;
      const float* xl = x + (bb * TLEN + ii) * LATENT;
#pragma unroll
      for (int j = 0; j < 8; ++j) HST[m * 64 + j] = (_Float16)er[j];
#pragma unroll
      for (int j = 0; j < 8; ++j) GIN[m * 96 + 64 + j] = (_Float16)xl[j];
#pragma unroll
      for (int j = 81; j < 96; ++j) GIN[m * 96 + j] = (_Float16)0.0f;
    } else {
      const float* xl = x + (bb * TLEN + ii + 1) * LATENT;
#pragma unroll
      for (int j = 8; j < 32; ++j) HST[m * 64 + j] = (_Float16)0.0f;
#pragma unroll
      for (int j = 0; j < 8; ++j) GIN[m * 96 + 72 + j] = (_Float16)xl[j];
    }
  }

  const int n   = lane & 15;
  const int m0b = (lane >> 4) << 3;

  {  // fc forward: 8->64, 64->64, 64->64, 64->64 (last no activation)
    v8f c[4] = {vzero8(), vzero8(), vzero8(), vzero8()};
    wave_gemm<1>(HST, 64, WT1, 96, c, lane);
#pragma unroll
    for (int nt = 0; nt < 4; ++nt)
#pragma unroll
      for (int r = 0; r < 8; ++r) {
        float p = c[nt][r] + sbias[nt * 16 + n];
        c[nt][r] = p >= 0.0f ? p : 0.1f * p;
      }
    stage_c(HST, c, lane);

#pragma unroll
    for (int L = 0; L < 2; ++L) {
      const _Float16* W = (L == 0) ? WT2 : WT3;
      v8f c2[4] = {vzero8(), vzero8(), vzero8(), vzero8()};
      wave_gemm<2>(HST, 64, W, 64, c2, lane);
#pragma unroll
      for (int nt = 0; nt < 4; ++nt)
#pragma unroll
        for (int r = 0; r < 8; ++r) {
          float p = c2[nt][r] + sbias[64 + L * 64 + nt * 16 + n];
          c2[nt][r] = p >= 0.0f ? p : 0.1f * p;
        }
      stage_c(HST, c2, lane);
    }

    v8f c4[4] = {vzero8(), vzero8(), vzero8(), vzero8()};
    wave_gemm<2>(HST, 64, WT4, 64, c4, lane);
    // final fc layer: +bias, no activation -> GIN cols 0..63
#pragma unroll
    for (int nt = 0; nt < 4; ++nt)
#pragma unroll
      for (int r = 0; r < 8; ++r)
        GIN[(m0b + r) * 96 + nt * 16 + n] = (_Float16)(c4[nt][r] + sbias[192 + nt * 16 + n]);
  }

  // ---------------- Phase G: per-dim flow MLP + JVP ----------------
  float logacc[8];
#pragma unroll
  for (int r = 0; r < 8; ++r) logacc[r] = 0.0f;

#pragma clang loop unroll(disable)
  for (int d = 0; d < LATENT; ++d) {
    __syncthreads();
    load_wt(gW1 + d * 81 * 64, WT1, 81, 96, tid);
    load_wt(gWh + (d * 2 + 0) * 4096, WT2, 64, 64, tid);
    load_wt(gWh + (d * 2 + 1) * 4096, WT3, 64, 64, tid);
    for (int idx = tid; idx < 16 * 64; idx += 128) {  // W2 (64->1): B = 64x16, col 0 live
      int nn = idx >> 6, kk = idx & 63;
      WT4[idx] = (nn == 0) ? (_Float16)gW2[d * 64 + kk] : (_Float16)0.0f;
    }
    load_bias(gb1 + d * 64, sbias, 64, tid);
    load_bias(gbh + d * 128, sbias + 64, 64, tid);
    load_bias(gbh + d * 128 + 64, sbias + 128, 64, tid);
    if (tid == 0) sbias[192] = gb2[d];
    __syncthreads();

    if (lane < 16) {  // x_t[:, d] -> GIN col 80 (wave-private)
      int s = s0 + lane;
      int bb = s / LEN;
      int ii = s - bb * LEN;
      GIN[lane * 96 + 80] = (_Float16)x[(bb * TLEN + ii + LAGS) * LATENT + d];
    }

    // L1: 81 (pad 96) -> 64; tangent through L1 is W1[80,:] * mask (no GEMM).
    v8f ch[4] = {vzero8(), vzero8(), vzero8(), vzero8()};
    wave_gemm<3>(GIN, 96, WT1, 96, ch, lane);
    v8f cd[4];
#pragma unroll
    for (int nt = 0; nt < 4; ++nt) {
      float d1 = (float)WT1[(nt * 16 + n) * 96 + 80];  // W1[80][col] from transposed store
#pragma unroll
      for (int r = 0; r < 8; ++r) {
        float p  = ch[nt][r] + sbias[nt * 16 + n];
        float mk = p >= 0.0f ? 1.0f : 0.1f;
        ch[nt][r] = p * mk;
        cd[nt][r] = d1 * mk;
      }
    }

    v16h ah0, ah1, ad0, ad1;
    stage_c(HST, ch, lane);
    ah0 = lds_load_a(HST, 64, 0, lane);
    ah1 = lds_load_a(HST, 64, 32, lane);
    stage_c(HST, cd, lane);
    ad0 = lds_load_a(HST, 64, 0, lane);
    ad1 = lds_load_a(HST, 64, 32, lane);

    // Hidden layers L2, L3: forward GEMM + tangent GEMM + leaky mask.
#pragma unroll
    for (int L = 0; L < 2; ++L) {
      const _Float16* W = (L == 0) ? WT2 : WT3;
      v8f h2[4] = {vzero8(), vzero8(), vzero8(), vzero8()};
      v8f d2[4] = {vzero8(), vzero8(), vzero8(), vzero8()};
      frag_gemm(ah0, ah1, W, 64, h2, lane);
      frag_gemm(ad0, ad1, W, 64, d2, lane);
#pragma unroll
      for (int nt = 0; nt < 4; ++nt)
#pragma unroll
        for (int r = 0; r < 8; ++r) {
          float p  = h2[nt][r] + sbias[64 + L * 64 + nt * 16 + n];
          float mk = p >= 0.0f ? 1.0f : 0.1f;
          h2[nt][r] = p * mk;
          d2[nt][r] = d2[nt][r] * mk;
        }
      stage_c(HST, h2, lane);
      ah0 = lds_load_a(HST, 64, 0, lane);
      ah1 = lds_load_a(HST, 64, 32, lane);
      stage_c(HST, d2, lane);
      ad0 = lds_load_a(HST, 64, 0, lane);
      ad1 = lds_load_a(HST, 64, 32, lane);
    }

    // Output layer 64->1 (WMMA against zero-padded 64x16 B, col 0 live).
    v8f co = vzero8(), cdo = vzero8();
    {
      v16h b0 = lds_load_bT(WT4, 64, 0, 0, lane);
      v16h b1 = lds_load_bT(WT4, 64, 32, 0, lane);
      co  = __builtin_amdgcn_wmma_f32_16x16x32_f16(false, ah0, false, b0, (short)0, co,  false, false);
      co  = __builtin_amdgcn_wmma_f32_16x16x32_f16(false, ah1, false, b1, (short)0, co,  false, false);
      cdo = __builtin_amdgcn_wmma_f32_16x16x32_f16(false, ad0, false, b0, (short)0, cdo, false, false);
      cdo = __builtin_amdgcn_wmma_f32_16x16x32_f16(false, ad1, false, b1, (short)0, cdo, false, false);
    }

    if (n == 0) {  // lanes 0 and 16 own column 0 for rows m0b..m0b+7
#pragma unroll
      for (int r = 0; r < 8; ++r) {
        int s = s0 + m0b + r;
        out[s * LATENT + d] = co[r] + sbias[192];
        logacc[r] += __logf(fabsf(cdo[r]));
      }
    }
  }

  if (n == 0) {
#pragma unroll
    for (int r = 0; r < 8; ++r) out[NSAMP * LATENT + s0 + m0b + r] = logacc[r];
  }
}

extern "C" void kernel_launch(void* const* d_in, const int* in_sizes, int n_in,
                              void* d_out, int out_size, void* d_ws, size_t ws_size,
                              hipStream_t stream) {
  (void)in_sizes; (void)n_in; (void)d_ws; (void)ws_size; (void)out_size;
  const float* x    = (const float*)d_in[0];
  const float* emb  = (const float*)d_in[1];
  const float* fcW1 = (const float*)d_in[2];
  const float* fcb1 = (const float*)d_in[3];
  const float* fcWh = (const float*)d_in[4];
  const float* fcbh = (const float*)d_in[5];
  const float* fcW2 = (const float*)d_in[6];
  const float* fcb2 = (const float*)d_in[7];
  const float* gW1  = (const float*)d_in[8];
  const float* gb1  = (const float*)d_in[9];
  const float* gWh  = (const float*)d_in[10];
  const float* gbh  = (const float*)d_in[11];
  const float* gW2  = (const float*)d_in[12];
  const float* gb2  = (const float*)d_in[13];
  float* out = (float*)d_out;

  dim3 grid(NSAMP / 64);  // 2046 blocks, 64 rows each
  dim3 block(128);        // 4 waves x 16 rows
  np_flow_kernel<<<grid, block, 0, stream>>>(x, emb, fcW1, fcb1, fcWh, fcbh, fcW2, fcb2,
                                             gW1, gb1, gWh, gbh, gW2, gb2, out);
}
